// LatentCountingRegisters_53661321396449
// MI455X (gfx1250) — matrix operand order repaired and checked by reading, not compiled, not run on hardware
//
#include <hip/hip_runtime.h>
#include <hip/hip_bf16.h>
#include <math.h>

// ---------------- problem constants ----------------
#define B_N   64
#define N_PIX 4096          // H*W
#define E_DIM 256
#define C_COL 10
#define NF_   8
#define NH_   8
#define DH_   32            // E/NH
#define CIN_  17            // 1 + 2*NF
#define S_SPL 8             // key splits per batch (kernel 2 grid = B*S)
#define CHUNK (N_PIX / S_SPL)   // 512 keys per block
#define KT    32                // keys per inner iteration
#define NIT   (CHUNK / KT)      // 16 iterations

typedef __bf16 bf16;
typedef __attribute__((ext_vector_type(16))) __bf16 v16bf;
typedef __attribute__((ext_vector_type(8)))  float  v8f;

struct U8x32 { uint4 a, b; };   // 32 bytes == one 16-element bf16 fragment slice

// LDS budget for kernel 2 (<= 320KB = 327,680B):
// wk+wv swizzled bf16 (256KB) + k,v swizzled tiles (32KB) + scores f32 (16KB)
// + P bf16 (8KB) + gcol (128B) + row stats (1.5KB) = 321,152 B
static constexpr size_t SMEM2 =
    (size_t)2 * E_DIM * E_DIM * sizeof(bf16) +
    (size_t)2 * KT * E_DIM * sizeof(bf16) +
    (size_t)8 * 16 * KT * sizeof(float) +
    (size_t)8 * 16 * KT * sizeof(bf16) +
    (size_t)KT * sizeof(int) +
    (size_t)8 * 48 * sizeof(float);

__device__ __forceinline__ void wave_lds_sync() {
  __builtin_amdgcn_wave_barrier();
  asm volatile("s_wait_dscnt 0" ::: "memory");
  __builtin_amdgcn_wave_barrier();
}

// contiguous 32B fragment load (lowers to ds_load_b128 x2 / global_load_b128 x2)
__device__ __forceinline__ v16bf load_frag32(const bf16* p) {
  U8x32 t;
  t.a = ((const uint4*)p)[0];
  t.b = ((const uint4*)p)[1];
  return __builtin_bit_cast(v16bf, t);
}

__device__ __forceinline__ float gelu_exact(float x) {
  return 0.5f * x * (1.0f + erff(x * 0.7071067811865475f));
}

__device__ __forceinline__ float block_sum256(float v, float* s_red) {
  int t = threadIdx.x;
  s_red[t] = v; __syncthreads();
  for (int off = 128; off > 0; off >>= 1) {
    if (t < off) s_red[t] += s_red[t + off];
    __syncthreads();
  }
  float r = s_red[0]; __syncthreads();
  return r;
}

// =====================================================================
// Kernel 1: histogram -> fourier enc -> count_embed (GELU+LN) -> q proj
// grid = B, block = 256
// =====================================================================
__global__ void lcr_count_embed_q(const int* __restrict__ grid,
                                  const float* __restrict__ cq,
                                  const float* __restrict__ w_count,
                                  const float* __restrict__ b_count,
                                  const float* __restrict__ g_ln1,
                                  const float* __restrict__ b_ln1,
                                  const float* __restrict__ wq,
                                  const float* __restrict__ bq,
                                  float* __restrict__ counts_out,
                                  float* __restrict__ count_embed_out,
                                  bf16* __restrict__ q_out) {
  __shared__ float s_cnt[C_COL];
  __shared__ float s_enc[C_COL][CIN_];
  __shared__ float s_emb[C_COL][E_DIM];
  __shared__ float s_red[256];

  const int b = blockIdx.x;
  const int t = threadIdx.x;

  if (t < C_COL) s_cnt[t] = 0.0f;
  __syncthreads();

  const int* grow = grid + (size_t)b * N_PIX;
  for (int i = t; i < N_PIX; i += 256) {
    int g = grow[i];
    g = g < 0 ? 0 : (g > C_COL - 1 ? C_COL - 1 : g);
    atomicAdd(&s_cnt[g], 1.0f);
  }
  __syncthreads();

  if (t < C_COL) counts_out[(size_t)b * C_COL + t] = s_cnt[t];

  for (int idx = t; idx < C_COL * CIN_; idx += 256) {
    int c = idx / CIN_, j = idx % CIN_;
    float cn = s_cnt[c] * (1.0f / (float)N_PIX);
    float v;
    if (j == 0) v = cn;
    else if (j <= NF_) v = sinf(cn * exp2f((float)(j - 1)) * 3.14159265358979323846f);
    else               v = cosf(cn * exp2f((float)(j - 1 - NF_)) * 3.14159265358979323846f);
    s_enc[c][j] = v;
  }
  __syncthreads();

  for (int idx = t; idx < C_COL * E_DIM; idx += 256) {
    int c = idx / E_DIM, e = idx % E_DIM;
    float acc = b_count[e];
#pragma unroll
    for (int j = 0; j < CIN_; ++j) acc += s_enc[c][j] * w_count[(size_t)j * E_DIM + e];
    s_emb[c][e] = gelu_exact(acc);
  }
  __syncthreads();

  for (int c = 0; c < C_COL; ++c) {
    float x = s_emb[c][t];
    float mean = block_sum256(x, s_red) * (1.0f / E_DIM);
    float d = x - mean;
    float var = block_sum256(d * d, s_red) * (1.0f / E_DIM);
    float y = d * rsqrtf(var + 1e-5f) * g_ln1[t] + b_ln1[t];
    s_emb[c][t] = y;
    count_embed_out[((size_t)b * C_COL + c) * E_DIM + t] = y;
    __syncthreads();
  }

  for (int idx = t; idx < 16 * E_DIM; idx += 256) {
    int row = idx / E_DIM, e = idx % E_DIM;
    float acc = 0.0f;
    if (row < C_COL) {
      acc = bq[e];
      for (int j = 0; j < E_DIM; ++j)
        acc += (cq[(size_t)row * E_DIM + j] + s_emb[row][j]) * wq[(size_t)j * E_DIM + e];
    }
    q_out[(size_t)b * 16 * E_DIM + idx] = (bf16)acc;
  }
}

// =====================================================================
// Kernel 2: fused K/V projection (WMMA bf16) + masked flash attention
// grid = B*S, block = 256 (8 waves; wave w == head w)
//
// All WMMA B operands read from LDS live in fragment-major layout:
//   [tile][lane(32)][16 bf16]  ->  one 32B contiguous read per lane.
// =====================================================================
__global__ void lcr_fused_attn(const float* __restrict__ features,
                               const int* __restrict__ grid,
                               const bf16* __restrict__ qbf,
                               const float* __restrict__ wk,
                               const float* __restrict__ bk,
                               const float* __restrict__ wv,
                               const float* __restrict__ bv,
                               float* __restrict__ part_m,
                               float* __restrict__ part_l,
                               float* __restrict__ part_acc) {
  extern __shared__ __align__(128) char smem[];
  // swizzled weights: [eo_t(16)*8 + ko][lane(32)][16]
  bf16*  s_wk = (bf16*)smem;                        // E*E
  bf16*  s_wv = s_wk + E_DIM * E_DIM;               // E*E
  // swizzled k: [head(8)*2 + ktile][lane(32)][16]  (K dim = dh of that head)
  bf16*  s_k  = s_wv + E_DIM * E_DIM;               // KT*E
  // swizzled v: [dtile(16)][lane(32)][16]          (K dim = 32 chunk keys)
  bf16*  s_v  = s_k + KT * E_DIM;                   // KT*E
  float* s_sc = (float*)(s_v + KT * E_DIM);         // [8][16][KT] scores
  bf16*  s_P  = (bf16*)(s_sc + 8 * 16 * KT);        // [8][16][KT] probs
  int*   s_gc = (int*)(s_P + 8 * 16 * KT);          // [KT]
  float* s_rs = (float*)(s_gc + KT);                // [8][48]: m, l, scale

  const int tid  = threadIdx.x;
  const int lane = tid & 31;
  const int wave = tid >> 5;                        // == head index
  const int b    = blockIdx.x / S_SPL;
  const int spl  = blockIdx.x % S_SPL;
  const int nb0  = spl * CHUNK;

  // ---- stage weights fp32 -> bf16 swizzled into fragment-major LDS ----
  // slot = (eo_t*8 + ko)*32 + lane_b ; element i: e = ko*32+(lane_b>>4)*16+i,
  // eo = eo_t*16 + (lane_b&15)
  for (int slot = tid; slot < 128 * 32; slot += 256) {
    const int p  = slot >> 5;
    const int l  = slot & 31;
    const int eb = (p & 7) * 32 + (l >> 4) * 16;
    const int eo = (p >> 3) * 16 + (l & 15);
    union { bf16 h[16]; uint4 q[2]; } tk, tv;
#pragma unroll
    for (int i = 0; i < 16; ++i) {
      tk.h[i] = (bf16)wk[(size_t)(eb + i) * E_DIM + eo];
      tv.h[i] = (bf16)wv[(size_t)(eb + i) * E_DIM + eo];
    }
    uint4* dk = (uint4*)(s_wk + (size_t)slot * 16);
    uint4* dv = (uint4*)(s_wv + (size_t)slot * 16);
    dk[0] = tk.q[0]; dk[1] = tk.q[1];
    dv[0] = tv.q[0]; dv[1] = tv.q[1];
  }

  // ---- this head's q A-fragment (16 queries x dh=32), two 16B loads ----
  v16bf qa;
  {
    const int row = lane & 15;
    const int hi  = (lane >> 4) * 8;
    const bf16* qrow = qbf + ((size_t)b * 16 + row) * E_DIM + wave * DH_;
    U8x32 t;
    t.a = *(const uint4*)(qrow + hi);
    t.b = *(const uint4*)(qrow + 16 + hi);
    qa = __builtin_bit_cast(v16bf, t);
  }

  float* mrow = s_rs + wave * 48;
  float* lrow = mrow + 16;
  float* srow = lrow + 16;
  if (lane < 16) { mrow[lane] = -3.0e38f; lrow[lane] = 0.0f; srow[lane] = 1.0f; }

  v8f oacc0 = {}; v8f oacc1 = {};
  __syncthreads();

  // projection split: wave -> (keytile, k-or-v, 8 eo tiles)
  const int kt_tile = wave >> 2;
  const int mat     = (wave >> 1) & 1;
  const int eobase  = (wave & 1) * 8;
  const bf16*  s_wsel = mat ? s_wv : s_wk;
  const float* bias   = mat ? bv   : bk;
  const float* fb = features + (size_t)b * E_DIM * N_PIX;

  for (int it = 0; it < NIT; ++it) {
    const int n0 = nb0 + it * KT;

    if (tid < KT) {
      int g = grid[(size_t)b * N_PIX + n0 + tid];
      s_gc[tid] = g < 0 ? 0 : (g > C_COL - 1 ? C_COL - 1 : g);
    }
    if (it + 1 < NIT)
      __builtin_prefetch(fb + (size_t)(n0 + KT) + (size_t)lane * N_PIX, 0, 1);

    // ---- K/V projection: 8 output tiles per wave, K loop over E ----
    v8f acc[8];
#pragma unroll
    for (int j = 0; j < 8; ++j) acc[j] = (v8f){};

    const int arow = n0 + kt_tile * 16 + (lane & 15);
    const int ahi  = (lane >> 4) * 16;
    for (int ko = 0; ko < 8; ++ko) {
      v16bf a;
      const float* fcol = fb + arow;
#pragma unroll
      for (int i = 0; i < 16; ++i) {
        int e = ko * 32 + ahi + i;
        a[i] = (bf16)fcol[(size_t)e * N_PIX];
      }
      const bf16* wbase = s_wsel + (size_t)(((eobase * 8) + ko) * 32 + lane) * 16;
#pragma unroll
      for (int j = 0; j < 8; ++j) {
        v16bf bm = load_frag32(wbase + (size_t)j * (8 * 32 * 16));
        acc[j] = __builtin_amdgcn_wmma_f32_16x16x32_bf16(
            false, a, false, bm, (short)0, acc[j], false, false);
      }
    }
    // ---- scatter C-fragments (+bias) into swizzled k/v tile layouts ----
    {
      const int rbase = (lane >> 4) * 8;          // C row offset within tile
      const int bcol  = lane & 15;
#pragma unroll
      for (int j = 0; j < 8; ++j) {
        const int eo  = (eobase + j) * 16 + bcol;
        const float bia = bias[eo];
        if (!mat) {
          // k swz: slot = (head*2 + kt_tile)*32 + ((dl>>3)&1)*16 + keycol ; elem = (dl&7) + ((dl&16)>>1)
          const int head = eo >> 5, dl = eo & 31;
          bf16* dst = s_k + (size_t)((head * 2 + kt_tile) * 32 + ((dl >> 3) & 1) * 16) * 16 + (dl & 7) + ((dl & 16) >> 1);
#pragma unroll
          for (int r = 0; r < 8; ++r) {
            const int keycol = rbase + r;         // 0..15
            dst[(size_t)keycol * 16] = (bf16)(acc[j][r] + bia);
          }
        } else {
          // v swz: slot = dtile*32 + ((key>>3)&1)*16 + dcol ; elem = (key&7) + ((key&16)>>1)
          const int dtile = eo >> 4, dcol = eo & 15;
          bf16* dst = s_v + (size_t)(dtile * 32) * 16 + dcol * 16;
#pragma unroll
          for (int r = 0; r < 8; ++r) {
            const int key = kt_tile * 16 + rbase + r;   // 0..31
            dst[(size_t)(((key >> 3) & 1) * 16) * 16 + (key & 7) + ((key & 16) >> 1)] = (bf16)(acc[j][r] + bia);
          }
        }
      }
    }
    __syncthreads();

    // ---- scores = q . k^T for this head (one WMMA per 16-key tile) ----
    float* sw = s_sc + wave * 16 * KT;   // [16][KT]
    {
      const int col = lane & 15;
      const int rr  = (lane >> 4) * 8;
      const float inv_sqrt_dh = 0.17677669529663687f;
#pragma unroll
      for (int ktile = 0; ktile < 2; ++ktile) {
        v16bf kb = load_frag32(s_k + (size_t)((wave * 2 + ktile) * 32 + lane) * 16);
        v8f scf = {};
        scf = __builtin_amdgcn_wmma_f32_16x16x32_bf16(
            false, qa, false, kb, (short)0, scf, false, false);
#pragma unroll
        for (int r = 0; r < 8; ++r)
          sw[(size_t)(rr + r) * KT + ktile * 16 + col] = scf[r] * inv_sqrt_dh;
      }
    }
    wave_lds_sync();

    // ---- masked online softmax: lanes 0..15 each own one query row ----
    if (lane < 16) {
      int c = lane;
      float rowv[KT];
#pragma unroll
      for (int j = 0; j < KT; ++j) rowv[j] = sw[(size_t)c * KT + j];
      float mo = mrow[c];
      float mx = -3.0e38f;
#pragma unroll
      for (int j = 0; j < KT; ++j)
        if (s_gc[j] == c) mx = fmaxf(mx, rowv[j]);
      float mn = fmaxf(mo, mx);
      float sc = __expf(mo - mn);
      float rs = 0.0f;
      bf16* pw = s_P + (size_t)wave * 16 * KT + (size_t)c * KT;
#pragma unroll
      for (int j = 0; j < KT; ++j) {
        float p = (s_gc[j] == c) ? __expf(rowv[j] - mn) : 0.0f;
        rs += p;
        pw[j] = (bf16)p;
      }
      mrow[c] = mn;
      lrow[c] = lrow[c] * sc + rs;
      srow[c] = sc;
    }
    wave_lds_sync();

    // rescale running output accumulators by per-row scale
    {
      int rr = (lane >> 4) * 8;
#pragma unroll
      for (int r = 0; r < 8; ++r) {
        float scl = srow[rr + r];
        oacc0[r] *= scl;
        oacc1[r] *= scl;
      }
    }

    // ---- P @ V : A = P (16 x 32 keys), B = v slices (32 x 16) ----
    v16bf pa;
    {
      const int row = lane & 15;
      const int hi  = (lane >> 4) * 8;
      const bf16* pr = s_P + ((size_t)wave * 16 + row) * KT;
      U8x32 t;
      t.a = *(const uint4*)(pr + hi);
      t.b = *(const uint4*)(pr + 16 + hi);
      pa = __builtin_bit_cast(v16bf, t);
    }
    {
      v16bf vb0 = load_frag32(s_v + (size_t)((wave * 2 + 0) * 32 + lane) * 16);
      v16bf vb1 = load_frag32(s_v + (size_t)((wave * 2 + 1) * 32 + lane) * 16);
      oacc0 = __builtin_amdgcn_wmma_f32_16x16x32_bf16(
          false, pa, false, vb0, (short)0, oacc0, false, false);
      oacc1 = __builtin_amdgcn_wmma_f32_16x16x32_bf16(
          false, pa, false, vb1, (short)0, oacc1, false, false);
    }
    __syncthreads();   // protect s_k/s_v/s_gc before next chunk overwrites
  }

  // ---- emit split partials: m, l, unnormalized acc ----
  {
    float* pm_w = part_m + ((size_t)blockIdx.x * NH_ + wave) * 16;
    float* pl_w = part_l + ((size_t)blockIdx.x * NH_ + wave) * 16;
    if (lane < 16) { pm_w[lane] = mrow[lane]; pl_w[lane] = lrow[lane]; }
    float* pa_out = part_acc + (size_t)blockIdx.x * 16 * E_DIM;
    int col = lane & 15;
    int rr  = (lane >> 4) * 8;
#pragma unroll
    for (int r = 0; r < 8; ++r) {
      pa_out[(size_t)(rr + r) * E_DIM + wave * DH_ + col]      = oacc0[r];
      pa_out[(size_t)(rr + r) * E_DIM + wave * DH_ + 16 + col] = oacc1[r];
    }
  }
}

// =====================================================================
// Kernel 3: combine splits (LSE merge) -> out proj -> concat -> w_out
//           -> GELU -> LN.  grid = B, block = 256
// =====================================================================
__global__ void lcr_combine_out(const float* __restrict__ counts,
                                const float* __restrict__ count_embed,
                                const float* __restrict__ part_m,
                                const float* __restrict__ part_l,
                                const float* __restrict__ part_acc,
                                const float* __restrict__ wo,
                                const float* __restrict__ bo,
                                const float* __restrict__ w_out,
                                const float* __restrict__ b_out,
                                const float* __restrict__ g_ln2,
                                const float* __restrict__ b_ln2,
                                float* __restrict__ out) {
  __shared__ float s_M[C_COL][NH_];
  __shared__ float s_L[C_COL][NH_];
  __shared__ float s_hc[C_COL];
  __shared__ float s_attn[C_COL][E_DIM];
  __shared__ float s_out[C_COL][E_DIM];
  __shared__ float s_red[256];

  const int b = blockIdx.x;
  const int t = threadIdx.x;

  if (t < C_COL * NH_) {
    int c = t / NH_, h = t % NH_;
    float M = -3.0e38f;
    for (int s = 0; s < S_SPL; ++s)
      M = fmaxf(M, part_m[(((size_t)b * S_SPL + s) * NH_ + h) * 16 + c]);
    float L = 0.0f;
    for (int s = 0; s < S_SPL; ++s) {
      float m = part_m[(((size_t)b * S_SPL + s) * NH_ + h) * 16 + c];
      float l = part_l[(((size_t)b * S_SPL + s) * NH_ + h) * 16 + c];
      L += l * __expf(m - M);
    }
    s_M[c][h] = M; s_L[c][h] = L;
  }
  if (t < C_COL) s_hc[t] = counts[(size_t)b * C_COL + t] > 0.5f ? 1.0f : 0.0f;
  __syncthreads();

  for (int idx = t; idx < C_COL * E_DIM; idx += 256) {
    int c = idx / E_DIM, e = idx % E_DIM, h = e / DH_;
    float M = s_M[c][h], L = s_L[c][h];
    float acc = 0.0f;
    for (int s = 0; s < S_SPL; ++s) {
      float m = part_m[(((size_t)b * S_SPL + s) * NH_ + h) * 16 + c];
      acc += part_acc[(((size_t)b * S_SPL + s) * 16 + c) * E_DIM + e] * __expf(m - M);
    }
    s_attn[c][e] = (L > 0.0f) ? acc / L : 0.0f;
  }
  __syncthreads();

  for (int idx = t; idx < C_COL * E_DIM; idx += 256) {
    int c = idx / E_DIM, e = idx % E_DIM;
    float acc = bo[e];
    for (int j = 0; j < E_DIM; ++j) acc += s_attn[c][j] * wo[(size_t)j * E_DIM + e];
    s_out[c][e] = acc * s_hc[c];
  }
  __syncthreads();

  for (int idx = t; idx < C_COL * E_DIM; idx += 256) {
    int c = idx / E_DIM, e = idx % E_DIM;
    const float* ce = count_embed + ((size_t)b * C_COL + c) * E_DIM;
    float acc = b_out[e];
    for (int j = 0; j < E_DIM; ++j) acc += ce[j] * w_out[(size_t)j * E_DIM + e];
    for (int j = 0; j < E_DIM; ++j) acc += s_out[c][j] * w_out[(size_t)(E_DIM + j) * E_DIM + e];
    s_attn[c][e] = gelu_exact(acc);
  }
  __syncthreads();

  for (int c = 0; c < C_COL; ++c) {
    float x = s_attn[c][t];
    float mean = block_sum256(x, s_red) * (1.0f / E_DIM);
    float d = x - mean;
    float var = block_sum256(d * d, s_red) * (1.0f / E_DIM);
    out[((size_t)b * C_COL + c) * E_DIM + t] = d * rsqrtf(var + 1e-5f) * g_ln2[t] + b_ln2[t];
    __syncthreads();
  }
}

// =====================================================================
extern "C" void kernel_launch(void* const* d_in, const int* in_sizes, int n_in,
                              void* d_out, int out_size, void* d_ws, size_t ws_size,
                              hipStream_t stream) {
  const int*   grid_in   = (const int*)d_in[0];
  const float* features  = (const float*)d_in[1];
  const float* cq        = (const float*)d_in[2];
  const float* w_count   = (const float*)d_in[3];
  const float* b_count   = (const float*)d_in[4];
  const float* g_ln1     = (const float*)d_in[5];
  const float* b_ln1     = (const float*)d_in[6];
  const float* wq        = (const float*)d_in[7];
  const float* bq        = (const float*)d_in[8];
  const float* wk        = (const float*)d_in[9];
  const float* bk        = (const float*)d_in[10];
  const float* wv        = (const float*)d_in[11];
  const float* bv        = (const float*)d_in[12];
  const float* wo        = (const float*)d_in[13];
  const float* bo        = (const float*)d_in[14];
  const float* w_out     = (const float*)d_in[15];
  const float* b_out     = (const float*)d_in[16];
  const float* g_ln2     = (const float*)d_in[17];
  const float* b_ln2     = (const float*)d_in[18];
  float* out = (float*)d_out;

  char* ws = (char*)d_ws;
  size_t off = 0;
  auto carve = [&](size_t bytes) {
    void* p = ws + off;
    off += (bytes + 255) & ~(size_t)255;
    return p;
  };
  float* counts      = (float*)carve((size_t)B_N * C_COL * sizeof(float));
  float* count_embed = (float*)carve((size_t)B_N * C_COL * E_DIM * sizeof(float));
  bf16*  qbf         = (bf16*) carve((size_t)B_N * 16 * E_DIM * sizeof(bf16));
  float* part_m      = (float*)carve((size_t)B_N * S_SPL * NH_ * 16 * sizeof(float));
  float* part_l      = (float*)carve((size_t)B_N * S_SPL * NH_ * 16 * sizeof(float));
  float* part_acc    = (float*)carve((size_t)B_N * S_SPL * 16 * E_DIM * sizeof(float));
  (void)ws_size; (void)in_sizes; (void)n_in; (void)out_size;

  hipFuncSetAttribute((const void*)lcr_fused_attn,
                      hipFuncAttributeMaxDynamicSharedMemorySize, (int)SMEM2);

  lcr_count_embed_q<<<B_N, 256, 0, stream>>>(
      grid_in, cq, w_count, b_count, g_ln1, b_ln1, wq, bq,
      counts, count_embed, qbf);

  lcr_fused_attn<<<B_N * S_SPL, 256, SMEM2, stream>>>(
      features, grid_in, qbf, wk, bk, wv, bv, part_m, part_l, part_acc);

  lcr_combine_out<<<B_N, 256, 0, stream>>>(
      counts, count_embed, part_m, part_l, part_acc,
      wo, bo, w_out, b_out, g_ln2, b_ln2, out);
}
